// MultiHeadAttention_59588376264899
// MI455X (gfx1250) — matrix operand order repaired and checked by reading, not compiled, not run on hardware
//
#include <hip/hip_runtime.h>
#include <hip/hip_bf16.h>

typedef __bf16 bf16_t;
typedef __attribute__((ext_vector_type(16))) __bf16 v16bf;
typedef __attribute__((ext_vector_type(8)))  __bf16 v8bf;
typedef __attribute__((ext_vector_type(4)))  __bf16 v4bf;
typedef __attribute__((ext_vector_type(8)))  float  v8f;

constexpr int kB = 2;
constexpr int kT = 2048;
constexpr int kC = 1024;
constexpr int kH = 16;
constexpr int kD = 64;
constexpr int kM = kB * kT;   // 4096 tokens total

// ---------------------------------------------------------------------------
// Fragment loaders (CDNA5 wave32 WMMA 16x16x32 bf16 layouts, ISA 7.12.2)
// A 16x32: lane holds row (lane&15); elems 0..7  = K[k0 + hi*8 .. +7]
//                                    elems 8..15 = K[k0 + 16 + hi*8 .. +7]
// B 32x16: lane holds col (lane&15); same K order as A
// ---------------------------------------------------------------------------
__device__ __forceinline__ v16bf load_a16(const bf16_t* rowptr, int k0, int hi) {
    v8bf lo = *(const v8bf*)(rowptr + k0 + hi * 8);
    v8bf hh = *(const v8bf*)(rowptr + k0 + 16 + hi * 8);
    v16bf r;
#pragma unroll
    for (int i = 0; i < 8; ++i) { r[i] = lo[i]; r[i + 8] = hh[i]; }
    return r;
}

__device__ __forceinline__ v16bf load_b16(const bf16_t* rowptr, int k0, int hi) {
    v8bf lo = *(const v8bf*)(rowptr + k0 + hi * 8);
    v8bf hh = *(const v8bf*)(rowptr + k0 + 16 + hi * 8);
    v16bf r;
#pragma unroll
    for (int i = 0; i < 8; ++i) { r[i] = lo[i]; r[i + 8] = hh[i]; }
    return r;
}

__device__ __forceinline__ v8f wmma_bf16(v16bf a, v16bf b, v8f c) {
    return __builtin_amdgcn_wmma_f32_16x16x32_bf16(
        /*neg_a=*/false, a, /*neg_b=*/false, b,
        /*c_mod=*/(short)0, c, /*reuse_a=*/false, /*reuse_b=*/false);
}

// ---------------------------------------------------------------------------
// fp32 -> bf16 conversion, 4 elements per thread
// ---------------------------------------------------------------------------
__global__ void cvt_kernel(const float* __restrict__ src, bf16_t* __restrict__ dst, int n) {
    int i = (blockIdx.x * blockDim.x + threadIdx.x) * 4;
    if (i < n) {
        float4 f = *(const float4*)(src + i);
        v4bf o;
        o[0] = (bf16_t)f.x; o[1] = (bf16_t)f.y; o[2] = (bf16_t)f.z; o[3] = (bf16_t)f.w;
        *(v4bf*)(dst + i) = o;
    }
}

// ---------------------------------------------------------------------------
// Fused QKV projection: Y = X @ W^T + b   (X:[4096,1024] bf16, W:[1024,1024])
// blockIdx.z selects {Q,K,V}. Q,K stored [BH][T][D]; V stored transposed
// [BH][D][T] so the attention PV B-fragment loads are contiguous.
// Block = 8 waves, 4(M) x 2(N); wave tile = 32 x 64; block tile = 128 x 128.
// K loop double-buffered: next-step fragments issued before current WMMAs.
// ---------------------------------------------------------------------------
__global__ __launch_bounds__(256) void qkv_gemm_kernel(
    const bf16_t* __restrict__ X,
    const bf16_t* __restrict__ Wqb, const bf16_t* __restrict__ Wkb, const bf16_t* __restrict__ Wvb,
    const float* __restrict__ bq, const float* __restrict__ bk, const float* __restrict__ bv,
    bf16_t* __restrict__ qh, bf16_t* __restrict__ kh, bf16_t* __restrict__ vt)
{
    const int lane = threadIdx.x & 31;
    const int wave = threadIdx.x >> 5;
    const int lr = lane & 15;
    const int hi = lane >> 4;
    const int mw = wave & 3;
    const int nw = wave >> 2;
    const int m0 = blockIdx.x * 128 + mw * 32;
    const int n0 = blockIdx.y * 128 + nw * 64;
    const int z  = blockIdx.z;

    const bf16_t* W    = (z == 0) ? Wqb : (z == 1) ? Wkb : Wvb;
    const float*  bias = (z == 0) ? bq  : (z == 1) ? bk  : bv;

    const bf16_t* arow0 = X + (size_t)(m0 + lr) * kC;
    const bf16_t* arow1 = X + (size_t)(m0 + 16 + lr) * kC;
    const bf16_t* brow  = W + (size_t)(n0 + lr) * kC;

    v8f acc0[4] = {}, acc1[4] = {};

    // ---- preload K-step 0 ----
    v16bf a0 = load_a16(arow0, 0, hi);
    v16bf a1 = load_a16(arow1, 0, hi);
    v16bf b0 = load_b16(brow,            0, hi);
    v16bf b1 = load_b16(brow + 16 * kC,  0, hi);
    v16bf b2 = load_b16(brow + 32 * kC,  0, hi);
    v16bf b3 = load_b16(brow + 48 * kC,  0, hi);

    for (int k0 = 32; k0 < kC; k0 += 32) {
        // issue next-step loads first (overlap with WMMAs below)
        v16bf a0n = load_a16(arow0, k0, hi);
        v16bf a1n = load_a16(arow1, k0, hi);
        v16bf b0n = load_b16(brow,           k0, hi);
        v16bf b1n = load_b16(brow + 16 * kC, k0, hi);
        v16bf b2n = load_b16(brow + 32 * kC, k0, hi);
        v16bf b3n = load_b16(brow + 48 * kC, k0, hi);

        acc0[0] = wmma_bf16(a0, b0, acc0[0]);  acc1[0] = wmma_bf16(a1, b0, acc1[0]);
        acc0[1] = wmma_bf16(a0, b1, acc0[1]);  acc1[1] = wmma_bf16(a1, b1, acc1[1]);
        acc0[2] = wmma_bf16(a0, b2, acc0[2]);  acc1[2] = wmma_bf16(a1, b2, acc1[2]);
        acc0[3] = wmma_bf16(a0, b3, acc0[3]);  acc1[3] = wmma_bf16(a1, b3, acc1[3]);

        a0 = a0n; a1 = a1n; b0 = b0n; b1 = b1n; b2 = b2n; b3 = b3n;
    }
    // tail K-step
    acc0[0] = wmma_bf16(a0, b0, acc0[0]);  acc1[0] = wmma_bf16(a1, b0, acc1[0]);
    acc0[1] = wmma_bf16(a0, b1, acc0[1]);  acc1[1] = wmma_bf16(a1, b1, acc1[1]);
    acc0[2] = wmma_bf16(a0, b2, acc0[2]);  acc1[2] = wmma_bf16(a1, b2, acc1[2]);
    acc0[3] = wmma_bf16(a0, b3, acc0[3]);  acc1[3] = wmma_bf16(a1, b3, acc1[3]);

    // ---- epilogue: bias add, scatter to head layouts ----
#pragma unroll
    for (int g = 0; g < 2; ++g) {
        const int mbase = m0 + g * 16;
#pragma unroll
        for (int j = 0; j < 4; ++j) {
#pragma unroll
            for (int v = 0; v < 8; ++v) {
                const float av = (g == 0) ? acc0[j][v] : acc1[j][v];
                const int mt = mbase + v + 8 * hi;     // token index
                const int c  = n0 + j * 16 + lr;       // output channel
                const float val = av + bias[c];
                const int bidx = mt >> 11;             // / kT
                const int t    = mt & (kT - 1);
                const int h    = c >> 6;               // / kD
                const int d    = c & (kD - 1);
                if (z < 2) {
                    bf16_t* dst = (z == 0) ? qh : kh;
                    dst[(((size_t)(bidx * kH + h)) * kT + t) * kD + d] = (bf16_t)val;
                } else {
                    vt[(((size_t)(bidx * kH + h)) * kD + d) * kT + t] = (bf16_t)val;
                }
            }
        }
    }
}

// ---------------------------------------------------------------------------
// Flash attention, causal. One wave owns 16 query rows; KV walked in chunks
// of 32 (= WMMA K). K-fragments double-buffered across chunks; V-fragments
// issued before the softmax so the exp/shuffle VALU work hides their latency.
// Block = 8 waves = 128 query rows. Grid = (T/128, B*H).
// ---------------------------------------------------------------------------
__global__ __launch_bounds__(256) void attn_kernel(
    const bf16_t* __restrict__ Q, const bf16_t* __restrict__ K,
    const bf16_t* __restrict__ VT, bf16_t* __restrict__ O)
{
    __shared__ __align__(16) bf16_t pls[8][16][32];   // 8 KB: per-wave P tile

    const int lane = threadIdx.x & 31;
    const int wave = threadIdx.x >> 5;
    const int lr = lane & 15;
    const int hi = lane >> 4;
    const int bh = blockIdx.y;
    const int q0 = blockIdx.x * 128 + wave * 16;

    const bf16_t* Qb = Q  + (size_t)bh * kT * kD;
    const bf16_t* Kb = K  + (size_t)bh * kT * kD;
    const bf16_t* Vb = VT + (size_t)bh * kD * kT;

    const bf16_t* qrow = Qb + (size_t)(q0 + lr) * kD;
    const v16bf qa0 = load_a16(qrow, 0, hi);
    const v16bf qa1 = load_a16(qrow, 32, hi);

    v8f oacc[4] = {};
    float mrow[8], lrow[8];
#pragma unroll
    for (int v = 0; v < 8; ++v) { mrow[v] = -1e30f; lrow[v] = 0.0f; }

    const int kvend = q0 + 16;

    // ---- preload K fragments for chunk kv0 = 0 ----
    v16bf kb0a = load_b16(Kb + (size_t)lr * kD,        0,  hi);
    v16bf kb0b = load_b16(Kb + (size_t)lr * kD,        32, hi);
    v16bf kb1a = load_b16(Kb + (size_t)(16 + lr) * kD, 0,  hi);
    v16bf kb1b = load_b16(Kb + (size_t)(16 + lr) * kD, 32, hi);

    for (int kv0 = 0; kv0 < kvend; kv0 += 32) {
        // ---- S tiles from pre-fetched K fragments ----
        v8f s0 = {}, s1 = {};
        s0 = wmma_bf16(qa0, kb0a, s0);  s0 = wmma_bf16(qa1, kb0b, s0);
        s1 = wmma_bf16(qa0, kb1a, s1);  s1 = wmma_bf16(qa1, kb1b, s1);

        // ---- issue this chunk's V fragments (consumed after softmax) ----
        v16bf vb0 = load_b16(Vb + (size_t)(lr) * kT,      kv0, hi);
        v16bf vb1 = load_b16(Vb + (size_t)(16 + lr) * kT, kv0, hi);
        v16bf vb2 = load_b16(Vb + (size_t)(32 + lr) * kT, kv0, hi);
        v16bf vb3 = load_b16(Vb + (size_t)(48 + lr) * kT, kv0, hi);

        // ---- issue next chunk's K fragments ----
        v16bf kb0an = kb0a, kb0bn = kb0b, kb1an = kb1a, kb1bn = kb1b;
        if (kv0 + 32 < kvend) {
            const bf16_t* nk0 = Kb + (size_t)(kv0 + 32 + lr) * kD;
            const bf16_t* nk1 = Kb + (size_t)(kv0 + 48 + lr) * kD;
            kb0an = load_b16(nk0, 0, hi);  kb0bn = load_b16(nk0, 32, hi);
            kb1an = load_b16(nk1, 0, hi);  kb1bn = load_b16(nk1, 32, hi);
        }

        const int tk0 = kv0 + lr;
        const int tk1 = kv0 + 16 + lr;

        // ---- online softmax per row (hides the in-flight loads) ----
#pragma unroll
        for (int v = 0; v < 8; ++v) {
            const int rg = q0 + v + 8 * hi;                   // global query row
            float a0 = (tk0 <= rg) ? s0[v] * 0.125f : -1e30f; // 1/sqrt(64)
            float a1 = (tk1 <= rg) ? s1[v] * 0.125f : -1e30f;
            float mx = fmaxf(a0, a1);
#pragma unroll
            for (int off = 1; off < 16; off <<= 1)
                mx = fmaxf(mx, __shfl_xor(mx, off, 32));
            const float mnew = fmaxf(mrow[v], mx);
            const float p0   = __expf(a0 - mnew);
            const float p1   = __expf(a1 - mnew);
            const float corr = __expf(mrow[v] - mnew);
            mrow[v] = mnew;
            float sm = p0 + p1;
#pragma unroll
            for (int off = 1; off < 16; off <<= 1)
                sm += __shfl_xor(sm, off, 32);
            lrow[v] = lrow[v] * corr + sm;
#pragma unroll
            for (int j = 0; j < 4; ++j) oacc[j][v] *= corr;
            // stage P (C-layout element -> row-major LDS tile)
            pls[wave][v + 8 * hi][lr]      = (bf16_t)p0;
            pls[wave][v + 8 * hi][16 + lr] = (bf16_t)p1;
        }

        // ---- O += P @ V  (P re-read in A-layout; V^T rows contiguous) ----
        const v16bf pa = load_a16(&pls[wave][lr][0], 0, hi);
        oacc[0] = wmma_bf16(pa, vb0, oacc[0]);
        oacc[1] = wmma_bf16(pa, vb1, oacc[1]);
        oacc[2] = wmma_bf16(pa, vb2, oacc[2]);
        oacc[3] = wmma_bf16(pa, vb3, oacc[3]);

        kb0a = kb0an; kb0b = kb0bn; kb1a = kb1an; kb1b = kb1bn;
    }

    // ---- normalize and store O [BH][T][D] ----
    bf16_t* Ob = O + (size_t)bh * kT * kD;
#pragma unroll
    for (int v = 0; v < 8; ++v) {
        const float inv = 1.0f / lrow[v];
        const int t = q0 + v + 8 * hi;
#pragma unroll
        for (int j = 0; j < 4; ++j)
            Ob[(size_t)t * kD + j * 16 + lr] = (bf16_t)(oacc[j][v] * inv);
    }
}

// ---------------------------------------------------------------------------
// Output projection: out = concat_heads(O) @ Wo^T + bo  (fp32 out)
// Same 32x64 double-buffered wave tile; A rows gathered from per-head O
// storage (32-wide K chunks never straddle a head boundary -> b128 loads).
// ---------------------------------------------------------------------------
__global__ __launch_bounds__(256) void oproj_kernel(
    const bf16_t* __restrict__ Oh, const bf16_t* __restrict__ Wob,
    const float* __restrict__ bo, float* __restrict__ out)
{
    const int lane = threadIdx.x & 31;
    const int wave = threadIdx.x >> 5;
    const int lr = lane & 15;
    const int hi = lane >> 4;
    const int mw = wave & 3;
    const int nw = wave >> 2;
    const int m0 = blockIdx.x * 128 + mw * 32;
    const int n0 = blockIdx.y * 128 + nw * 64;

    const int mt0 = m0 + lr;            // lane's A row, group 0
    const int mt1 = m0 + 16 + lr;       // lane's A row, group 1
    const int b0i = mt0 >> 11, t0 = mt0 & (kT - 1);
    const int b1i = mt1 >> 11, t1 = mt1 & (kT - 1);
    const bf16_t* brow = Wob + (size_t)(n0 + lr) * kC;

    v8f acc0[4] = {}, acc1[4] = {};

    auto arow = [&](int bidx, int t, int k0) -> const bf16_t* {
        const int h  = k0 >> 6;            // head of this K chunk
        const int db = k0 & (kD - 1);      // 0 or 32
        return Oh + (((size_t)(bidx * kH + h)) * kT + t) * kD + db;
    };

    // ---- preload K-step 0 ----
    v16bf a0 = load_a16(arow(b0i, t0, 0), 0, hi);
    v16bf a1 = load_a16(arow(b1i, t1, 0), 0, hi);
    v16bf b0 = load_b16(brow,           0, hi);
    v16bf b1 = load_b16(brow + 16 * kC, 0, hi);
    v16bf b2 = load_b16(brow + 32 * kC, 0, hi);
    v16bf b3 = load_b16(brow + 48 * kC, 0, hi);

    for (int k0 = 32; k0 < kC; k0 += 32) {
        v16bf a0n = load_a16(arow(b0i, t0, k0), 0, hi);
        v16bf a1n = load_a16(arow(b1i, t1, k0), 0, hi);
        v16bf b0n = load_b16(brow,           k0, hi);
        v16bf b1n = load_b16(brow + 16 * kC, k0, hi);
        v16bf b2n = load_b16(brow + 32 * kC, k0, hi);
        v16bf b3n = load_b16(brow + 48 * kC, k0, hi);

        acc0[0] = wmma_bf16(a0, b0, acc0[0]);  acc1[0] = wmma_bf16(a1, b0, acc1[0]);
        acc0[1] = wmma_bf16(a0, b1, acc0[1]);  acc1[1] = wmma_bf16(a1, b1, acc1[1]);
        acc0[2] = wmma_bf16(a0, b2, acc0[2]);  acc1[2] = wmma_bf16(a1, b2, acc1[2]);
        acc0[3] = wmma_bf16(a0, b3, acc0[3]);  acc1[3] = wmma_bf16(a1, b3, acc1[3]);

        a0 = a0n; a1 = a1n; b0 = b0n; b1 = b1n; b2 = b2n; b3 = b3n;
    }
    acc0[0] = wmma_bf16(a0, b0, acc0[0]);  acc1[0] = wmma_bf16(a1, b0, acc1[0]);
    acc0[1] = wmma_bf16(a0, b1, acc0[1]);  acc1[1] = wmma_bf16(a1, b1, acc1[1]);
    acc0[2] = wmma_bf16(a0, b2, acc0[2]);  acc1[2] = wmma_bf16(a1, b2, acc1[2]);
    acc0[3] = wmma_bf16(a0, b3, acc0[3]);  acc1[3] = wmma_bf16(a1, b3, acc1[3]);

#pragma unroll
    for (int g = 0; g < 2; ++g) {
        const int mbase = m0 + g * 16;
#pragma unroll
        for (int j = 0; j < 4; ++j) {
#pragma unroll
            for (int v = 0; v < 8; ++v) {
                const float av = (g == 0) ? acc0[j][v] : acc1[j][v];
                const int mt = mbase + v + 8 * hi;
                const int c  = n0 + j * 16 + lr;
                out[(size_t)mt * kC + c] = av + bo[c];
            }
        }
    }
}

// ---------------------------------------------------------------------------
// Host-side launch
// ---------------------------------------------------------------------------
extern "C" void kernel_launch(void* const* d_in, const int* in_sizes, int n_in,
                              void* d_out, int out_size, void* d_ws, size_t ws_size,
                              hipStream_t stream) {
    (void)in_sizes; (void)n_in; (void)out_size; (void)ws_size;
    const float* x  = (const float*)d_in[0];
    const float* Wq = (const float*)d_in[1];
    const float* bq = (const float*)d_in[2];
    const float* Wk = (const float*)d_in[3];
    const float* bk = (const float*)d_in[4];
    const float* Wv = (const float*)d_in[5];
    const float* bv = (const float*)d_in[6];
    const float* Wo = (const float*)d_in[7];
    const float* bo = (const float*)d_in[8];
    float* out = (float*)d_out;

    // workspace carve-up (bf16): x, 4 weights, q, k, v^T, o  => ~48 MB
    bf16_t* p   = (bf16_t*)d_ws;
    bf16_t* xb  = p; p += (size_t)kM * kC;
    bf16_t* wqb = p; p += (size_t)kC * kC;
    bf16_t* wkb = p; p += (size_t)kC * kC;
    bf16_t* wvb = p; p += (size_t)kC * kC;
    bf16_t* wob = p; p += (size_t)kC * kC;
    bf16_t* qh  = p; p += (size_t)kM * kC;
    bf16_t* kh  = p; p += (size_t)kM * kC;
    bf16_t* vt  = p; p += (size_t)kM * kC;
    bf16_t* oh  = p; p += (size_t)kM * kC;

    const int nx = kM * kC;     // 4,194,304
    const int nw = kC * kC;     // 1,048,576
    cvt_kernel<<<(nx / 4 + 255) / 256, 256, 0, stream>>>(x,  xb,  nx);
    cvt_kernel<<<(nw / 4 + 255) / 256, 256, 0, stream>>>(Wq, wqb, nw);
    cvt_kernel<<<(nw / 4 + 255) / 256, 256, 0, stream>>>(Wk, wkb, nw);
    cvt_kernel<<<(nw / 4 + 255) / 256, 256, 0, stream>>>(Wv, wvb, nw);
    cvt_kernel<<<(nw / 4 + 255) / 256, 256, 0, stream>>>(Wo, wob, nw);

    qkv_gemm_kernel<<<dim3(kM / 128, kC / 128, 3), 256, 0, stream>>>(
        xb, wqb, wkb, wvb, bq, bk, bv, qh, kh, vt);

    attn_kernel<<<dim3(kT / 128, kB * kH), 256, 0, stream>>>(qh, kh, vt, oh);

    oproj_kernel<<<dim3(kM / 128, kC / 128), 256, 0, stream>>>(oh, wob, bo, out);
}
